// MultiHeadAttention_69106023793143
// MI455X (gfx1250) — matrix operand order repaired and checked by reading, not compiled, not run on hardware
//
#include <hip/hip_runtime.h>
#include <hip/hip_bf16.h>
#include <math.h>

#define BB 4
#define SS 2048
#define DD 1024
#define HH 16
#define HDIM 64
#define BH (BB*HH)

typedef __bf16 bf16;
typedef __attribute__((ext_vector_type(16))) __bf16 v16bf;
typedef __attribute__((ext_vector_type(8)))  __bf16 v8bf;
typedef __attribute__((ext_vector_type(8)))  float  v8f;
typedef __attribute__((ext_vector_type(4)))  float  v4f;
typedef __attribute__((ext_vector_type(4)))  unsigned int u32x4;
typedef __attribute__((ext_vector_type(8)))  int    i32x8;
typedef __attribute__((ext_vector_type(4)))  int    i32x4;

#if defined(__has_builtin)
#if __has_builtin(__builtin_amdgcn_tensor_load_to_lds)
#define HAVE_TDM 1
#endif
#endif
#ifndef HAVE_TDM
#define HAVE_TDM 0
#endif

#if __has_include(<hip/amd_detail/amd_gfx1250_TDM.h>)
#define TDM_SIX_ARGS 1
#else
#define TDM_SIX_ARGS 0
#endif

static __device__ __forceinline__ v16bf cat16(v8bf lo, v8bf hi) {
  return __builtin_shufflevector(lo, hi, 0,1,2,3,4,5,6,7,8,9,10,11,12,13,14,15);
}
static __device__ __forceinline__ v8f wmma_bf16(v16bf a, v16bf b, v8f c) {
  // D = A(16x32 bf16) * B(32x16 bf16) + C(16x16 f32)
  return __builtin_amdgcn_wmma_f32_16x16x32_bf16(false, a, false, b, (short)0, c, false, false);
}

#if HAVE_TDM
// LDS byte offset of a generic pointer into __shared__ space.
static __device__ __forceinline__ unsigned lds_offset_of(const void* p) {
  return (unsigned)(unsigned long)(__attribute__((address_space(3))) const void*)p;
}
// TDM: async DMA a 2D tile (rows x cols, 2-byte elems, row stride in elems)
// from global memory into LDS at byte offset ldsOff. Tracked by TENSORcnt.
static __device__ __forceinline__ void tdm_load_tile_bf16(
    const bf16* gsrc, unsigned ldsOff, unsigned rows, unsigned cols,
    unsigned rowStrideElems) {
  unsigned long long ga = (unsigned long long)(const void*)gsrc;
  u32x4 g0;
  g0[0] = 1u;                                        // count=1, user mode
  g0[1] = ldsOff;                                    // lds_addr (bytes)
  g0[2] = (unsigned)(ga & 0xFFFFFFFFu);              // global_addr[31:0]
  g0[3] = (unsigned)((ga >> 32) & 0x1FFFFFFu) | (2u << 30);  // addr[56:32] | type=2
  i32x8 g1;
  g1[0] = (int)(1u << 16);                           // workgroup_mask=0, data_size=2B
  g1[1] = (int)((cols & 0xFFFFu) << 16);             // tensor_dim0[15:0]
  g1[2] = (int)((cols >> 16) | ((rows & 0xFFFFu) << 16));  // dim0[31:16] | dim1[15:0]
  g1[3] = (int)((rows >> 16) | (cols << 16));        // dim1[31:16] | tile_dim0
  g1[4] = (int)rows;                                 // tile_dim1 | tile_dim2=0
  g1[5] = (int)rowStrideElems;                       // tensor_dim0_stride[31:0]
  g1[6] = 0;                                         // stride0[47:32] | stride1[15:0]
  g1[7] = 0;                                         // stride1[47:16]
  i32x4 gz = {0, 0, 0, 0};
#if TDM_SIX_ARGS
  i32x8 gz8 = {0, 0, 0, 0, 0, 0, 0, 0};
  __builtin_amdgcn_tensor_load_to_lds(g0, g1, gz, gz, gz8, 0);
#else
  __builtin_amdgcn_tensor_load_to_lds(g0, g1, gz, gz, 0);
#endif
}
#endif  // HAVE_TDM

// ---------------------------------------------------------------------------
// Projection: Out[b,h,s,hd] (bf16) = (X[(b,s),:] @ W + bias) head-split.
// 128x64 block tile, 8 waves x (2x2 WMMA 16x16 tiles), K stepped by 32.
// ---------------------------------------------------------------------------
__global__ __launch_bounds__(256) void proj_qkv_kernel(
    const float* __restrict__ X, const float* __restrict__ W,
    const float* __restrict__ bias, bf16* __restrict__ Out) {
  __shared__ __align__(16) bf16 Xs[128 * 32];  // [m][k]
  __shared__ __align__(16) bf16 Ws[64 * 32];   // [n][k] (transposed)
  const int tid  = threadIdx.x;
  const int lane = tid & 31;
  const int w    = tid >> 5;
  const int m0   = blockIdx.y * 128;
  const int n0   = blockIdx.x * 64;
  const int wm   = (w >> 1) * 32;
  const int wn   = (w & 1) * 32;
  const int hh   = lane >> 4;
  const int ln   = lane & 15;

  v8f acc[2][2] = {};

  for (int k0 = 0; k0 < DD; k0 += 32) {
    {
      const int row = tid >> 1, cb = (tid & 1) * 16;
      const float* src = X + (size_t)(m0 + row) * DD + k0 + cb;
      #pragma unroll
      for (int i = 0; i < 16; i += 4) {
        v4f f = *(const v4f*)(src + i);
        Xs[row * 32 + cb + i + 0] = (bf16)f.x;
        Xs[row * 32 + cb + i + 1] = (bf16)f.y;
        Xs[row * 32 + cb + i + 2] = (bf16)f.z;
        Xs[row * 32 + cb + i + 3] = (bf16)f.w;
      }
    }
    {
      const int k = tid >> 3, nb = (tid & 7) * 8;
      const float* src = W + (size_t)(k0 + k) * DD + n0 + nb;
      #pragma unroll
      for (int j = 0; j < 8; ++j) Ws[(nb + j) * 32 + k] = (bf16)src[j];
    }
    __syncthreads();

    v16bf a[2], bb[2];
    #pragma unroll
    for (int rt = 0; rt < 2; ++rt) {
      const int m = wm + rt * 16 + ln;
      a[rt] = cat16(*(const v8bf*)&Xs[m * 32 + 8 * hh],
                    *(const v8bf*)&Xs[m * 32 + 16 + 8 * hh]);
    }
    #pragma unroll
    for (int ct = 0; ct < 2; ++ct) {
      const int n = wn + ct * 16 + ln;
      bb[ct] = cat16(*(const v8bf*)&Ws[n * 32 + 8 * hh],
                     *(const v8bf*)&Ws[n * 32 + 16 + 8 * hh]);
    }
    #pragma unroll
    for (int rt = 0; rt < 2; ++rt)
      #pragma unroll
      for (int ct = 0; ct < 2; ++ct)
        acc[rt][ct] = wmma_bf16(a[rt], bb[ct], acc[rt][ct]);
    __syncthreads();
  }

  #pragma unroll
  for (int ct = 0; ct < 2; ++ct) {
    const int N  = n0 + wn + ct * 16 + ln;
    const float bv = bias[N];
    const int h = N >> 6, hd = N & 63;
    #pragma unroll
    for (int rt = 0; rt < 2; ++rt) {
      #pragma unroll
      for (int r = 0; r < 8; ++r) {
        const int M = m0 + wm + rt * 16 + r + 8 * hh;
        const int b = M >> 11, s = M & 2047;
        Out[(size_t)((b * HH + h) * SS + s) * HDIM + hd] = (bf16)(acc[rt][ct][r] + bv);
      }
    }
  }
}

// ---------------------------------------------------------------------------
// Fused sigmoid attention, double-buffered K/V staging. K tiles arrive via the
// Tensor Data Mover (async, TENSORcnt) when available; V is software-pipelined
// with a transpose-on-store. O += sigmoid(Q K^T / 8) V, scores never spilled.
// ---------------------------------------------------------------------------
__global__ __launch_bounds__(256) void sigmoid_attn_kernel(
    const bf16* __restrict__ Qh, const bf16* __restrict__ Kh,
    const bf16* __restrict__ Vh, float* __restrict__ Out) {
  __shared__ __align__(16) bf16 Ks[2][32 * 64];     // [buf][key][d]
  __shared__ __align__(16) bf16 Vt[2][64 * 32];     // [buf][e][key] (transposed)
  __shared__ __align__(16) bf16 Pw[8 * 16 * 32];    // per-wave prob tile [m][j]
  const int tid = threadIdx.x, lane = tid & 31, w = tid >> 5;
  const int hh = lane >> 4, ln = lane & 15;
  const int bh = blockIdx.y;
  const int b = bh >> 4, h = bh & 15;
  const int sw = blockIdx.x * 128 + w * 16;
  const size_t bhBase = (size_t)bh * SS * HDIM;

  const int skey = tid >> 3, seb = (tid & 7) * 8;   // staging coords

  // persistent Q A-fragments (16x32 for d=[0,32) and [32,64))
  v16bf qa[2];
  #pragma unroll
  for (int t = 0; t < 2; ++t) {
    const bf16* qrow = Qh + bhBase + (size_t)(sw + ln) * HDIM + t * 32 + 8 * hh;
    qa[t] = cat16(*(const v8bf*)qrow, *(const v8bf*)(qrow + 16));
  }

  v8f oacc[4] = {};
  bf16* Pme = &Pw[w * 16 * 32];

  // ---- staging helpers (inlined twice: prologue + steady state) ----
  #define STAGE_V(BUF, KB)                                                   \
    do {                                                                     \
      const v8bf vv = *(const v8bf*)(Vh + bhBase +                           \
                                     (size_t)((KB) + skey) * HDIM + seb);    \
      _Pragma("unroll")                                                      \
      for (int j = 0; j < 8; ++j) Vt[BUF][(seb + j) * 32 + skey] = vv[j];    \
    } while (0)

#if HAVE_TDM
  #define STAGE_K(BUF, KB)                                                   \
    do {                                                                     \
      if (w == 0)                                                            \
        tdm_load_tile_bf16(Kh + bhBase + (size_t)(KB) * HDIM,                \
                           lds_offset_of(&Ks[BUF][0]), 32u, 64u, 64u);       \
    } while (0)
#else
  #define STAGE_K(BUF, KB)                                                   \
    do {                                                                     \
      const v8bf kv = *(const v8bf*)(Kh + bhBase +                           \
                                     (size_t)((KB) + skey) * HDIM + seb);    \
      *(v8bf*)&Ks[BUF][skey * 64 + seb] = kv;                                \
    } while (0)
#endif

  // prologue: stage buffer 0
  STAGE_K(0, 0);
  STAGE_V(0, 0);

  for (int kb = 0; kb < SS; kb += 32) {
    const int buf = (kb >> 5) & 1;
    const bool more = (kb + 32) < SS;
    if (more) {            // prefetch next block into the other buffer
      STAGE_K(buf ^ 1, kb + 32);
      STAGE_V(buf ^ 1, kb + 32);
    }
#if HAVE_TDM
    if (w == 0) {          // drain current tile's TDM, keep prefetch in flight
      if (more) __builtin_amdgcn_s_wait_tensorcnt(1);
      else      __builtin_amdgcn_s_wait_tensorcnt(0);
    }
#endif
    __syncthreads();

    // scores (2 x 16x16 tiles over 32 keys), sigmoid, stash bf16 A-source
    #pragma unroll
    for (int jt = 0; jt < 2; ++jt) {
      v8f c = {};
      #pragma unroll
      for (int dh = 0; dh < 2; ++dh) {
        const bf16* kr = &Ks[buf][(jt * 16 + ln) * 64 + dh * 32 + 8 * hh];
        v16bf bfrag = cat16(*(const v8bf*)kr, *(const v8bf*)(kr + 16));
        c = wmma_bf16(qa[dh], bfrag, c);
      }
      #pragma unroll
      for (int r = 0; r < 8; ++r) {
        const float x = c[r] * 0.125f;                 // 1/sqrt(64)
        const float p = 1.0f / (1.0f + __expf(-x));
        Pme[(r + 8 * hh) * 32 + jt * 16 + ln] = (bf16)p;
      }
    }
    // per-wave LDS RAW: DS ops are in-order within a wave; fence compiler
    asm volatile("s_wait_dscnt 0" ::: "memory");

    // reload probs as A-fragment, accumulate O += P @ V
    const bf16* pr = &Pme[ln * 32 + 8 * hh];
    v16bf pfrag = cat16(*(const v8bf*)pr, *(const v8bf*)(pr + 16));
    #pragma unroll
    for (int et = 0; et < 4; ++et) {
      const bf16* vr = &Vt[buf][(et * 16 + ln) * 32 + 8 * hh];
      v16bf vfrag = cat16(*(const v8bf*)vr, *(const v8bf*)(vr + 16));
      oacc[et] = wmma_bf16(pfrag, vfrag, oacc[et]);
    }
    __syncthreads();
  }

  // epilogue: merge heads -> Out[B,S,D] fp32
  #pragma unroll
  for (int et = 0; et < 4; ++et) {
    const int col = h * 64 + et * 16 + ln;
    #pragma unroll
    for (int r = 0; r < 8; ++r) {
      const int s = sw + r + 8 * hh;
      Out[(size_t)(b * SS + s) * DD + col] = oacc[et][r];
    }
  }
  #undef STAGE_V
  #undef STAGE_K
}

extern "C" void kernel_launch(void* const* d_in, const int* in_sizes, int n_in,
                              void* d_out, int out_size, void* d_ws, size_t ws_size,
                              hipStream_t stream) {
  (void)in_sizes; (void)n_in; (void)out_size; (void)ws_size;
  const size_t per = (size_t)BH * SS * HDIM;   // elements per projected stream
  bf16* qws = (bf16*)d_ws;
  bf16* kws = qws + per;
  bf16* vws = kws + per;
  bf16* outs[3] = {qws, kws, vws};

  dim3 pgrid(DD / 64, (BB * SS) / 128, 1);
  for (int z = 0; z < 3; ++z) {
    proj_qkv_kernel<<<pgrid, 256, 0, stream>>>(
        (const float*)d_in[z],          // q / k / v
        (const float*)d_in[3 + 2 * z],  // Wq / Wk / Wv
        (const float*)d_in[4 + 2 * z],  // bq / bk / bv
        outs[z]);
  }

  dim3 agrid(SS / 128, BH, 1);
  sigmoid_attn_kernel<<<agrid, 256, 0, stream>>>(qws, kws, vws, (float*)d_out);
}
